// Classifier_16415365005684
// MI455X (gfx1250) — matrix operand order repaired and checked by reading, not compiled, not run on hardware
//
#include <hip/hip_runtime.h>

typedef __attribute__((ext_vector_type(16))) _Float16 v16h;
typedef __attribute__((ext_vector_type(8)))  float    v8f;

constexpr int NN = 32768;     // nodes
constexpr int EE = 262144;    // edges
constexpr int NLIG = 1024;

// ---------- WMMA helpers (C layout per CDNA5 ISA 7.12.2; A/B use contiguous k = 16h+s) ----------

__device__ __forceinline__ v8f wmma32(v16h a, v16h b, v8f c) {
  return __builtin_amdgcn_wmma_f32_16x16x32_f16(false, a, false, b, (short)0, c,
                                                false, false);
}

// A 16x32 f16 tile from LDS (row-major, leading dim `ld` in halves).
// lane<16: K = {k0+0..15}; lane>=16: K = {k0+16..31}
__device__ __forceinline__ v16h load_a_frag(const _Float16* As, int ld, int k0, int lane) {
  int g = lane >> 4, m = lane & 15;
  const _Float16* p = As + m * ld + k0 + g * 16;
  v16h a;
#pragma unroll
  for (int i = 0; i < 8; i++) a[i] = p[i];
#pragma unroll
  for (int i = 0; i < 8; i++) a[8 + i] = p[8 + i];
  return a;
}

// B 32x16 f16 fragment from transposed f16 weights Bt[n][k] (leading dim ld=Kpad).
// lane (g,n): element i -> K = k0 + g*16 + i, col = nb*16+n  => 32 contiguous bytes.
__device__ __forceinline__ v16h load_b_frag(const _Float16* Bt, int ld, int k0, int nb, int lane) {
  int g = lane >> 4, n = lane & 15;
  return *(const v16h*)(Bt + (size_t)(nb * 16 + n) * ld + k0 + g * 16);
}

__device__ __forceinline__ float fsigmoid(float x) { return 1.f / (1.f + __expf(-x)); }
__device__ __forceinline__ float fsilu(float x)    { return x * fsigmoid(x); }

// reduce across the 16 lanes of a half-wave group (rows are group-local in C layout)
__device__ __forceinline__ float group_sum(float v) {
#pragma unroll
  for (int m = 1; m < 16; m <<= 1) v += __shfl_xor(v, m, 32);
  return v;
}

// ---------- weight preparation ----------

// dst[n*Kpad + k] = (k < Ksrc) ? src[k*Ncols + n] : 0   (f32 -> f16, transpose+pad)
__global__ void transpose_f16_kernel(const float* __restrict__ src, _Float16* __restrict__ dst,
                                     int Ksrc, int Kpad, int Ncols) {
  int idx = blockIdx.x * blockDim.x + threadIdx.x;
  if (idx >= Ncols * Kpad) return;
  int n = idx / Kpad, k = idx % Kpad;
  dst[idx] = (k < Ksrc) ? (_Float16)src[k * Ncols + n] : (_Float16)0.f;
}

// v[f] = sum_j W_oe[f][j]*W_f[j];  c = b_oe . W_f + b_f
__global__ void head_prep_kernel(const float* __restrict__ W_oe, const float* __restrict__ b_oe,
                                 const float* __restrict__ W_f, const float* __restrict__ b_f,
                                 float* __restrict__ v, float* __restrict__ c) {
  int f = threadIdx.x;  // 128 threads
  float s = 0.f;
  for (int j = 0; j < 64; j++) s += W_oe[f * 64 + j] * W_f[j];
  v[f] = s;
  if (f == 0) {
    float cc = b_f[0];
    for (int j = 0; j < 64; j++) cc += b_oe[j] * W_f[j];
    *c = cc;
  }
}

// ---------- edge_attr: temb[t_bond[sbi]] (16) ++ gaussian smear (20), pad->64 f16 ----------

__global__ void edge_attr_kernel(const float* __restrict__ x, const int* __restrict__ edges,
                                 const int* __restrict__ t_bond, const float* __restrict__ temb,
                                 _Float16* __restrict__ attr) {
  int e = blockIdx.x * blockDim.x + threadIdx.x;
  if (e >= EE) return;
  int row = edges[e], col = edges[EE + e];
  float dx = x[row * 3 + 0] - x[col * 3 + 0];
  float dy = x[row * 3 + 1] - x[col * 3 + 1];
  float dz = x[row * 3 + 2] - x[col * 3 + 2];
  float dist = fminf(sqrtf(dx * dx + dy * dy + dz * dz), 4.f);
  int sbi = row * 31 + col - (row >> 5) * 32 - (row < col ? 1 : 0);
  int tb = t_bond[sbi];
  _Float16* o = attr + (size_t)e * 64;
#pragma unroll
  for (int j = 0; j < 16; j++) o[j] = (_Float16)temb[tb * 16 + j];
  const float LOG5 = 1.6094379124341003f;
  float offprev = 0.f;
#pragma unroll
  for (int j = 0; j < 20; j++) {
    float offj = __expf(LOG5 * (float)j * (1.f / 19.f)) - 1.f;
    float d = (j == 0) ? (__expf(LOG5 * (1.f / 19.f)) - 1.f) : (offj - offprev);
    float coeff = -0.5f / (d * d);
    float dd = dist - offj;
    o[16 + j] = (_Float16)__expf(coeff * dd * dd);
    offprev = offj;
  }
#pragma unroll
  for (int j = 36; j < 64; j++) o[j] = (_Float16)0.f;
}

// ---------- init: hh = concat(h, temb[t]) @ W_in + b_in  (K=32, one WMMA step) ----------

__global__ __launch_bounds__(64) void init_h_kernel(const float* __restrict__ h,
                                                    const int* __restrict__ t,
                                                    const float* __restrict__ temb,
                                                    const _Float16* __restrict__ Wint,
                                                    const float* __restrict__ b_in,
                                                    float* __restrict__ hh) {
  __shared__ _Float16 As[2][16 * 32];
  int wave = threadIdx.x >> 5, lane = threadIdx.x & 31;
  int n0 = (blockIdx.x * 2 + wave) * 16;
  _Float16* A = As[wave];
  for (int i = lane; i < 16 * 32; i += 32) {
    int r = i >> 5, c = i & 31, node = n0 + r;
    float val = (c < 16) ? h[node * 16 + c] : temb[t[node] * 16 + (c - 16)];
    A[i] = (_Float16)val;
  }
  __syncthreads();
  int g = lane >> 4, n = lane & 15;
  v16h a = load_a_frag(A, 32, 0, lane);
  v8f zero = {0.f, 0.f, 0.f, 0.f, 0.f, 0.f, 0.f, 0.f};
#pragma unroll
  for (int nb = 0; nb < 8; nb++) {
    v8f acc = wmma32(a, load_b_frag(Wint, 32, 0, nb, lane), zero);
    float bv = b_in[nb * 16 + n];
#pragma unroll
    for (int r = 0; r < 8; r++)
      hh[(size_t)(n0 + 8 * g + r) * 128 + nb * 16 + n] = acc[r] + bv;
  }
}

// ---------- edge kernel: m->MLP1(LN,SiLU)->MLP2(SiLU)->att gate->segment-sum ----------

__global__ __launch_bounds__(64) void edge_kernel(const float* __restrict__ hh,
                                                  const int* __restrict__ edges,
                                                  const _Float16* __restrict__ attr,
                                                  const _Float16* __restrict__ We1t,
                                                  const float* __restrict__ be1,
                                                  const float* __restrict__ g1,
                                                  const float* __restrict__ bt1,
                                                  const _Float16* __restrict__ We2t,
                                                  const float* __restrict__ be2,
                                                  const float* __restrict__ Watt,
                                                  const float* __restrict__ batt,
                                                  float* __restrict__ agg) {
  __shared__ _Float16 As[2][16 * 320];
  __shared__ int RC[2][32];
  int wave = threadIdx.x >> 5, lane = threadIdx.x & 31;
  int tile = blockIdx.x * 2 + wave;   // EE/16 tiles; 16 edges => nodes 2t, 2t+1
  int e0 = tile * 16;
  _Float16* A = As[wave];
  int* rc = RC[wave];
  if (lane < 16) {
    rc[lane]      = edges[e0 + lane];        // row
    rc[16 + lane] = edges[EE + e0 + lane];   // col
  }
  __syncthreads();
  // stage m = [hh[row] | hh[col] | edge_attr(pad 64)]  -> 16 x 320 f16
  for (int i = lane; i < 16 * 128; i += 32) {
    int r = i >> 7, c = i & 127;
    A[r * 320 + c]       = (_Float16)hh[(size_t)rc[r] * 128 + c];
    A[r * 320 + 128 + c] = (_Float16)hh[(size_t)rc[16 + r] * 128 + c];
  }
  for (int i = lane; i < 16 * 64; i += 32) {
    int r = i >> 6, c = i & 63;
    A[r * 320 + 256 + c] = attr[(size_t)(e0 + r) * 64 + c];
  }
  __syncthreads();

  int g = lane >> 4, n = lane & 15;
  v8f zero = {0.f, 0.f, 0.f, 0.f, 0.f, 0.f, 0.f, 0.f};
  v8f acc[8];
#pragma unroll
  for (int nb = 0; nb < 8; nb++) acc[nb] = zero;
#pragma unroll
  for (int ks = 0; ks < 10; ks++) {
    v16h a = load_a_frag(A, 320, ks * 32, lane);
#pragma unroll
    for (int nb = 0; nb < 8; nb++)
      acc[nb] = wmma32(a, load_b_frag(We1t, 320, ks * 32, nb, lane), acc[nb]);
  }
  // per-lane column params (cols nb*16+n)
  float be1v[8], g1v[8], bt1v[8], be2v[8], wattv[8];
#pragma unroll
  for (int nb = 0; nb < 8; nb++) {
    int cix = nb * 16 + n;
    be1v[nb] = be1[cix]; g1v[nb] = g1[cix]; bt1v[nb] = bt1[cix];
    be2v[nb] = be2[cix]; wattv[nb] = Watt[cix];
  }
  // LayerNorm(+be1) -> SiLU, write back as f16 A tile (stride 128)
#pragma unroll
  for (int r = 0; r < 8; r++) {
    float s = 0.f, ss = 0.f;
#pragma unroll
    for (int nb = 0; nb < 8; nb++) {
      float v0 = acc[nb][r] + be1v[nb];
      s += v0; ss += v0 * v0;
    }
    s = group_sum(s); ss = group_sum(ss);
    float mu = s * (1.f / 128.f);
    float var = ss * (1.f / 128.f) - mu * mu;
    float rstd = rsqrtf(var + 1e-5f);
    int rowi = g * 8 + r;
#pragma unroll
    for (int nb = 0; nb < 8; nb++) {
      float u = (acc[nb][r] + be1v[nb] - mu) * rstd * g1v[nb] + bt1v[nb];
      A[rowi * 128 + nb * 16 + n] = (_Float16)fsilu(u);
    }
  }
  __syncthreads();
  // GEMM2 (K=128)
  v8f acc2[8];
#pragma unroll
  for (int nb = 0; nb < 8; nb++) acc2[nb] = zero;
#pragma unroll
  for (int ks = 0; ks < 4; ks++) {
    v16h a = load_a_frag(A, 128, ks * 32, lane);
#pragma unroll
    for (int nb = 0; nb < 8; nb++)
      acc2[nb] = wmma32(a, load_b_frag(We2t, 128, ks * 32, nb, lane), acc2[nb]);
  }
  // mij = silu(. + be2); att = sigmoid(mij.Watt + batt); ef = mij*att
  float att[8];
#pragma unroll
  for (int r = 0; r < 8; r++) {
    float p = 0.f;
#pragma unroll
    for (int nb = 0; nb < 8; nb++) {
      float u = fsilu(acc2[nb][r] + be2v[nb]);
      acc2[nb][r] = u;
      p += u * wattv[nb];
    }
    p = group_sum(p);
    att[r] = fsigmoid(p + batt[0]);
  }
  // rows 8g..8g+7 belong to node 2*tile+g  => segment-sum in registers, /NORM_FACTOR
  int node = tile * 2 + g;
#pragma unroll
  for (int nb = 0; nb < 8; nb++) {
    float s2 = 0.f;
#pragma unroll
    for (int r = 0; r < 8; r++) s2 += acc2[nb][r] * att[r];
    agg[(size_t)node * 128 + nb * 16 + n] = s2 * 0.2f;
  }
}

// ---------- node kernel: concat(hh,agg)->MLP1(LN,SiLU)->MLP2 -> residual ----------

__global__ __launch_bounds__(64) void node_kernel(float* __restrict__ hh,
                                                  const float* __restrict__ agg,
                                                  const _Float16* __restrict__ Wn1t,
                                                  const float* __restrict__ bn1,
                                                  const float* __restrict__ g2,
                                                  const float* __restrict__ bt2,
                                                  const _Float16* __restrict__ Wn2t,
                                                  const float* __restrict__ bn2) {
  __shared__ _Float16 As[2][16 * 256];
  int wave = threadIdx.x >> 5, lane = threadIdx.x & 31;
  int n0 = (blockIdx.x * 2 + wave) * 16;
  _Float16* A = As[wave];
  for (int i = lane; i < 16 * 128; i += 32) {
    int r = i >> 7, c = i & 127;
    A[r * 256 + c]       = (_Float16)hh[(size_t)(n0 + r) * 128 + c];
    A[r * 256 + 128 + c] = (_Float16)agg[(size_t)(n0 + r) * 128 + c];
  }
  __syncthreads();
  int g = lane >> 4, n = lane & 15;
  v8f zero = {0.f, 0.f, 0.f, 0.f, 0.f, 0.f, 0.f, 0.f};
  v8f acc[8];
#pragma unroll
  for (int nb = 0; nb < 8; nb++) acc[nb] = zero;
#pragma unroll
  for (int ks = 0; ks < 8; ks++) {
    v16h a = load_a_frag(A, 256, ks * 32, lane);
#pragma unroll
    for (int nb = 0; nb < 8; nb++)
      acc[nb] = wmma32(a, load_b_frag(Wn1t, 256, ks * 32, nb, lane), acc[nb]);
  }
  float bn1v[8], g2v[8], bt2v[8], bn2v[8];
#pragma unroll
  for (int nb = 0; nb < 8; nb++) {
    int cix = nb * 16 + n;
    bn1v[nb] = bn1[cix]; g2v[nb] = g2[cix]; bt2v[nb] = bt2[cix]; bn2v[nb] = bn2[cix];
  }
#pragma unroll
  for (int r = 0; r < 8; r++) {
    float s = 0.f, ss = 0.f;
#pragma unroll
    for (int nb = 0; nb < 8; nb++) {
      float v0 = acc[nb][r] + bn1v[nb];
      s += v0; ss += v0 * v0;
    }
    s = group_sum(s); ss = group_sum(ss);
    float mu = s * (1.f / 128.f);
    float rstd = rsqrtf(ss * (1.f / 128.f) - mu * mu + 1e-5f);
    int rowi = g * 8 + r;
#pragma unroll
    for (int nb = 0; nb < 8; nb++) {
      float u = (acc[nb][r] + bn1v[nb] - mu) * rstd * g2v[nb] + bt2v[nb];
      A[rowi * 128 + nb * 16 + n] = (_Float16)fsilu(u);
    }
  }
  __syncthreads();
  v8f acc2[8];
#pragma unroll
  for (int nb = 0; nb < 8; nb++) acc2[nb] = zero;
#pragma unroll
  for (int ks = 0; ks < 4; ks++) {
    v16h a = load_a_frag(A, 128, ks * 32, lane);
#pragma unroll
    for (int nb = 0; nb < 8; nb++)
      acc2[nb] = wmma32(a, load_b_frag(Wn2t, 128, ks * 32, nb, lane), acc2[nb]);
  }
#pragma unroll
  for (int nb = 0; nb < 8; nb++) {
#pragma unroll
    for (int r = 0; r < 8; r++) {
      size_t idx = (size_t)(n0 + 8 * g + r) * 128 + nb * 16 + n;
      hh[idx] = hh[idx] + acc2[nb][r] + bn2v[nb];
    }
  }
}

// ---------- final head: out[lig] = mean_nodes(hh . v) + c ----------

__global__ void pool_kernel(const float* __restrict__ hh, const float* __restrict__ v,
                            const float* __restrict__ c, float* __restrict__ out) {
  int lig = blockIdx.x * (blockDim.x >> 5) + (threadIdx.x >> 5);
  int lane = threadIdx.x & 31;
  if (lig >= NLIG) return;
  const float* base = hh + (size_t)(lig * 32 + lane) * 128;
  float s = 0.f;
  for (int f = 0; f < 128; f++) s += base[f] * v[f];
#pragma unroll
  for (int m = 1; m < 32; m <<= 1) s += __shfl_xor(s, m, 32);
  if (lane == 0) out[lig] = s * (1.f / 32.f) + c[0];
}

// ---------- host launcher ----------

extern "C" void kernel_launch(void* const* d_in, const int* in_sizes, int n_in,
                              void* d_out, int out_size, void* d_ws, size_t ws_size,
                              hipStream_t stream) {
  (void)in_sizes; (void)n_in; (void)out_size; (void)ws_size;
  const float* x     = (const float*)d_in[0];
  const float* h     = (const float*)d_in[1];
  const int*   t     = (const int*)d_in[2];
  const int*   edges = (const int*)d_in[3];
  const int*   t_bond= (const int*)d_in[4];
  const float* temb  = (const float*)d_in[8];
  const float* W_in  = (const float*)d_in[9];
  const float* b_in  = (const float*)d_in[10];
  const float* We1   = (const float*)d_in[11];
  const float* be1   = (const float*)d_in[12];
  const float* g1    = (const float*)d_in[13];
  const float* bt1   = (const float*)d_in[14];
  const float* We2   = (const float*)d_in[15];
  const float* be2   = (const float*)d_in[16];
  const float* Watt  = (const float*)d_in[17];
  const float* batt  = (const float*)d_in[18];
  const float* Wn1   = (const float*)d_in[19];
  const float* bn1   = (const float*)d_in[20];
  const float* g2    = (const float*)d_in[21];
  const float* bt2   = (const float*)d_in[22];
  const float* Wn2   = (const float*)d_in[23];
  const float* bn2   = (const float*)d_in[24];
  const float* W_oe  = (const float*)d_in[25];
  const float* b_oe  = (const float*)d_in[26];
  const float* W_f   = (const float*)d_in[27];
  const float* b_f   = (const float*)d_in[28];

  char* ws = (char*)d_ws;
  float*    hh   = (float*)(ws + (size_t)0);            // NN*128*4  = 16 MB
  float*    agg  = (float*)(ws + (size_t)16777216);     // NN*128*4  = 16 MB
  _Float16* attr = (_Float16*)(ws + (size_t)33554432);  // EE*64*2   = 32 MB
  char* wb = ws + (size_t)67108864;                     // f16 weights
  _Float16* Wint = (_Float16*)wb;                       // 128*32
  const size_t LAYER_BYTES = 212992;                    // 81920+32768+65536+32768
  float* vhead = (float*)(wb + 8192 + 4 * LAYER_BYTES);
  float* chead = vhead + 128;

  // weight conversion (transpose f32 -> f16, pad K)
  transpose_f16_kernel<<<(128 * 32 + 255) / 256, 256, 0, stream>>>(W_in, Wint, 32, 32, 128);
  for (int l = 0; l < 4; l++) {
    char* lb = wb + 8192 + (size_t)l * LAYER_BYTES;
    _Float16* We1t = (_Float16*)(lb);
    _Float16* We2t = (_Float16*)(lb + 81920);
    _Float16* Wn1t = (_Float16*)(lb + 114688);
    _Float16* Wn2t = (_Float16*)(lb + 180224);
    transpose_f16_kernel<<<(128 * 320 + 255) / 256, 256, 0, stream>>>(We1 + (size_t)l * 292 * 128, We1t, 292, 320, 128);
    transpose_f16_kernel<<<(128 * 128 + 255) / 256, 256, 0, stream>>>(We2 + (size_t)l * 128 * 128, We2t, 128, 128, 128);
    transpose_f16_kernel<<<(128 * 256 + 255) / 256, 256, 0, stream>>>(Wn1 + (size_t)l * 256 * 128, Wn1t, 256, 256, 128);
    transpose_f16_kernel<<<(128 * 128 + 255) / 256, 256, 0, stream>>>(Wn2 + (size_t)l * 128 * 128, Wn2t, 128, 128, 128);
  }
  head_prep_kernel<<<1, 128, 0, stream>>>(W_oe, b_oe, W_f, b_f, vhead, chead);
  edge_attr_kernel<<<EE / 256, 256, 0, stream>>>(x, edges, t_bond, temb, attr);
  init_h_kernel<<<NN / 32, 64, 0, stream>>>(h, t, temb, Wint, b_in, hh);

  for (int l = 0; l < 4; l++) {
    char* lb = wb + 8192 + (size_t)l * LAYER_BYTES;
    _Float16* We1t = (_Float16*)(lb);
    _Float16* We2t = (_Float16*)(lb + 81920);
    _Float16* Wn1t = (_Float16*)(lb + 114688);
    _Float16* Wn2t = (_Float16*)(lb + 180224);
    edge_kernel<<<EE / 32, 64, 0, stream>>>(hh, edges, attr, We1t,
                                            be1 + l * 128, g1 + l * 128, bt1 + l * 128,
                                            We2t, be2 + l * 128, Watt + l * 128, batt + l, agg);
    node_kernel<<<NN / 32, 64, 0, stream>>>(hh, agg, Wn1t,
                                            bn1 + l * 128, g2 + l * 128, bt2 + l * 128,
                                            Wn2t, bn2 + l * 128);
  }
  pool_kernel<<<NLIG / 8, 256, 0, stream>>>(hh, vhead, chead, (float*)d_out);
}
